// Attention_33998961115529
// MI455X (gfx1250) — matrix operand order repaired and checked by reading, not compiled, not run on hardware
//
#include <hip/hip_runtime.h>
#include <cstdint>

// Problem constants (B=2, H=16, S=2048, D=64)
#define BH      32
#define S_LEN   2048
#define D_DIM   64
#define BLOCK_M 64
#define BLOCK_N 32
#define NWAVES  4
#define NITER   (S_LEN / BLOCK_N)

typedef __attribute__((ext_vector_type(16))) __bf16 v16bf;
typedef __attribute__((ext_vector_type(8)))  float  v8f;

union FragB {
    uint4 q[2];
    v16bf v;
};
union Bytes16 {
    uint4 u;
    unsigned char b[16];
};

// Native bf16 converts (v_cvt_pk_bf16_f32 class)
__device__ __forceinline__ unsigned int pack2(float lo, float hi) {
    union { __bf16 h[2]; unsigned int u; } c;
    c.h[0] = (__bf16)lo;
    c.h[1] = (__bf16)hi;
    return c.u;
}
__device__ __forceinline__ unsigned short f2bf(float f) {
    union { __bf16 h; unsigned short u; } c;
    c.h = (__bf16)f;
    return c.u;
}

// Max over the 16-lane row via DPP ROW_ROR (VALU only -- no LDS/bpermute).
// After ror8,ror4,ror2,ror1 every lane of the row holds the full max.
__device__ __forceinline__ float rowmax16(float x) {
    int v;
    v = __float_as_int(x);
    x = fmaxf(x, __int_as_float(__builtin_amdgcn_update_dpp(0, v, 0x128, 0xF, 0xF, true))); // ror:8
    v = __float_as_int(x);
    x = fmaxf(x, __int_as_float(__builtin_amdgcn_update_dpp(0, v, 0x124, 0xF, 0xF, true))); // ror:4
    v = __float_as_int(x);
    x = fmaxf(x, __int_as_float(__builtin_amdgcn_update_dpp(0, v, 0x122, 0xF, 0xF, true))); // ror:2
    v = __float_as_int(x);
    x = fmaxf(x, __int_as_float(__builtin_amdgcn_update_dpp(0, v, 0x121, 0xF, 0xF, true))); // ror:1
    return x;
}

__global__ __launch_bounds__(128)
void fa_fwd_kernel(const float* __restrict__ Q,
                   const float* __restrict__ K,
                   const float* __restrict__ V,
                   const unsigned char* __restrict__ mask,
                   float* __restrict__ out)
{
    // Double-buffered LDS tiles (24 KB total, WGP has 320 KB)
    __shared__ alignas(16) unsigned short Kt[2][BLOCK_N][D_DIM];      // [n][k]  bf16  8KB
    __shared__ alignas(16) unsigned short Vt[2][D_DIM][BLOCK_N];      // [d][n]  bf16  8KB
    __shared__ alignas(16) unsigned char  Mt[2][BLOCK_N][BLOCK_M];    // [n][qm] bool  4KB
    __shared__ alignas(16) unsigned short Pw[NWAVES][16][BLOCK_N];    //         bf16  4KB

    const int tid  = threadIdx.x;
    const int wave = tid >> 5;
    const int lane = tid & 31;
    const int l16  = lane & 15;
    const int hl   = lane >> 4;        // which 16-lane half of the wave

    const int bh     = blockIdx.y;
    const int m_base = blockIdx.x * BLOCK_M;
    const int q_base = m_base + wave * 16;   // this wave's 16 query rows

    const float* Qb = Q + (size_t)bh * S_LEN * D_DIM;
    const float* Kb = K + (size_t)bh * S_LEN * D_DIM;
    const float* Vb = V + (size_t)bh * S_LEN * D_DIM;
    float*       Ob = out + (size_t)bh * S_LEN * D_DIM;

    // K staging: 128 threads cover 32x64 f32, 16 floats each (one row segment)
    const int sn = tid >> 2;          // tile row 0..31
    const int sc = (tid & 3) * 16;    // col start 0,16,32,48
    // V staging: pair of key rows x 8 d-values -> packed dword transpose stores
    const int vnp = (tid >> 3) * 2;   // key-row pair base 0,2,..,30
    const int vdg = (tid & 7) * 8;    // d group start 0,8,..,56
    // mask staging: 2 threads per query row, 16 bytes each
    const int mq  = tid >> 1;         // query row within block 0..63
    const int mko = (tid & 1) * 16;   // key-col start 0 or 16

    // ---- Q A-fragments (resident whole kernel) ----
    // 16-bit A 16x32 layout: lane<16 -> M=lane, K={kb..kb+7, kb+16..kb+23}, kb = hl*8
    FragB qa[2];   // K-dim ranges [0,32) and [32,64)
    {
        const float* qrow = Qb + (size_t)(q_base + l16) * D_DIM;
        const int kb = hl * 8;
        #pragma unroll
        for (int g = 0; g < 2; ++g) {
            #pragma unroll
            for (int c = 0; c < 2; ++c) {
                const float* p = qrow + g * 32 + c * 16 + kb;
                float4 f0 = *(const float4*)(p);
                float4 f1 = *(const float4*)(p + 4);
                uint4 u;
                u.x = pack2(f0.x, f0.y); u.y = pack2(f0.z, f0.w);
                u.z = pack2(f1.x, f1.y); u.w = pack2(f1.z, f1.w);
                qa[g].q[c] = u;
            }
        }
    }

    // ones-column B fragment: B[k][0] = 1, everything else 0 (rowsum via WMMA)
    FragB ones;
    {
        const unsigned int w = (l16 == 0) ? 0x3F803F80u : 0u;  // bf16(1.0) pair
        ones.q[0] = make_uint4(w, w, w, w);
        ones.q[1] = make_uint4(w, w, w, w);
    }

    // tile loader (global -> registers) and converter (registers -> LDS buf)
    auto load_tile = [&](int kvb, float4 kr[4], float4 vr[4], uint4 &mreg) {
        const float* krow = Kb + (size_t)(kvb + sn) * D_DIM + sc;
        kr[0] = *(const float4*)(krow);
        kr[1] = *(const float4*)(krow + 4);
        kr[2] = *(const float4*)(krow + 8);
        kr[3] = *(const float4*)(krow + 12);
        const float* vrow0 = Vb + (size_t)(kvb + vnp) * D_DIM + vdg;
        vr[0] = *(const float4*)(vrow0);
        vr[1] = *(const float4*)(vrow0 + 4);
        vr[2] = *(const float4*)(vrow0 + D_DIM);
        vr[3] = *(const float4*)(vrow0 + D_DIM + 4);
        mreg = *(const uint4*)(mask + (size_t)(m_base + mq) * S_LEN + kvb + mko);
    };
    auto store_tile = [&](int buf, const float4 kr[4], const float4 vr[4], uint4 mreg) {
        uint4 u0, u1;
        u0.x = pack2(kr[0].x, kr[0].y); u0.y = pack2(kr[0].z, kr[0].w);
        u0.z = pack2(kr[1].x, kr[1].y); u0.w = pack2(kr[1].z, kr[1].w);
        u1.x = pack2(kr[2].x, kr[2].y); u1.y = pack2(kr[2].z, kr[2].w);
        u1.z = pack2(kr[3].x, kr[3].y); u1.w = pack2(kr[3].z, kr[3].w);
        *(uint4*)&Kt[buf][sn][sc]     = u0;
        *(uint4*)&Kt[buf][sn][sc + 8] = u1;

        // transposed V: dword = bf16 pair from two consecutive key rows, same d
        float a0[8] = {vr[0].x, vr[0].y, vr[0].z, vr[0].w,
                       vr[1].x, vr[1].y, vr[1].z, vr[1].w};
        float a1[8] = {vr[2].x, vr[2].y, vr[2].z, vr[2].w,
                       vr[3].x, vr[3].y, vr[3].z, vr[3].w};
        #pragma unroll
        for (int j = 0; j < 8; ++j)
            *(unsigned int*)&Vt[buf][vdg + j][vnp] = pack2(a0[j], a1[j]);

        Bytes16 mb; mb.u = mreg;
        #pragma unroll
        for (int j = 0; j < 16; ++j) Mt[buf][mko + j][mq] = mb.b[j];
    };

    // online-softmax state; C-layout: VGPR r -> row M = r + 8*hl, col N = l16
    float m_run[8];
    v8f acc[4];
    v8f acc_l;  // col 0 = running rowsum of exp (ones-column trick)
    const v8f vzero = {0.f,0.f,0.f,0.f,0.f,0.f,0.f,0.f};
    #pragma unroll
    for (int r = 0; r < 8; ++r) m_run[r] = -3.0e38f;
    #pragma unroll
    for (int f = 0; f < 4; ++f) acc[f] = vzero;
    acc_l = vzero;

    // work in log2 domain: softmax(a) == 2^(a*log2e) / sum(...)
    const float scl2 = 0.022097086912079608f * 1.4426950408889634f; // (1/sqrt(2048))*log2(e)
    const float MBIG = 1e10f * scl2;   // mask fill of -1e10 happens pre-scale (as reference)

    // ---- prologue: stage tile 0 into buffer 0 ----
    {
        float4 kr[4], vr[4]; uint4 mreg;
        load_tile(0, kr, vr, mreg);
        store_tile(0, kr, vr, mreg);
    }
    __syncthreads();

    for (int it = 0; it < NITER; ++it) {
        const int buf = it & 1;
        const int kv  = it * BLOCK_N;
        const bool have_next = (it + 1 < NITER);

        // issue next tile's global loads early (latency overlaps compute)
        float4 krn[4], vrn[4]; uint4 mrn;
        if (have_next) {
            load_tile(kv + BLOCK_N, krn, vrn, mrn);
            if (it + 2 < NITER) {   // gfx1250 global_prefetch_b8, 2 tiles ahead
                __builtin_prefetch(Kb + (size_t)(kv + 2 * BLOCK_N + sn) * D_DIM + sc, 0, 1);
                __builtin_prefetch(Vb + (size_t)(kv + 2 * BLOCK_N + vnp) * D_DIM + vdg, 0, 1);
                __builtin_prefetch(mask + (size_t)(m_base + mq) * S_LEN + kv + 2 * BLOCK_N + mko, 0, 1);
            }
        }

        // ---- S = Q * K^T  (two 16x16 score tiles over n; K-depth 64 = 2 WMMA) ----
        v8f s[2];
        #pragma unroll
        for (int t = 0; t < 2; ++t) {
            v8f c = vzero;
            #pragma unroll
            for (int g = 0; g < 2; ++g) {
                FragB kf;
                const uint4* p = (const uint4*)&Kt[buf][t * 16 + l16][g * 32 + hl * 8];
                kf.q[0] = p[0];
                kf.q[1] = p[2];
                c = __builtin_amdgcn_wmma_f32_16x16x32_bf16(
                        false, qa[g].v, false, kf.v, (short)0, c, false, false);
            }
            s[t] = c;
        }

        // ---- mask bytes for this lane: 8 rows @ fixed col = contiguous in Mt ----
        const unsigned long long mbits0 =
            *(const unsigned long long*)&Mt[buf][l16][wave * 16 + hl * 8];
        const unsigned long long mbits1 =
            *(const unsigned long long*)&Mt[buf][16 + l16][wave * 16 + hl * 8];

        // ---- scale + mask (arithmetic, mask byte is 0/1) + online max (DPP) ----
        #pragma unroll
        for (int r = 0; r < 8; ++r) {
            const float fm0 = (float)((unsigned int)(mbits0 >> (8 * r)) & 0xFFu);
            const float fm1 = (float)((unsigned int)(mbits1 >> (8 * r)) & 0xFFu);
            const float s0v = fmaf(fm0, -MBIG, s[0][r] * scl2);
            const float s1v = fmaf(fm1, -MBIG, s[1][r] * scl2);

            const float mx    = rowmax16(fmaxf(s0v, s1v));
            const float m_new = fmaxf(m_run[r], mx);
            const float alpha = exp2f(m_run[r] - m_new);
            const float e0 = exp2f(s0v - m_new);
            const float e1 = exp2f(s1v - m_new);
            m_run[r] = m_new;
            #pragma unroll
            for (int f = 0; f < 4; ++f) acc[f][r] *= alpha;
            acc_l[r] *= alpha;

            Pw[wave][hl * 8 + r][l16]      = f2bf(e0);
            Pw[wave][hl * 8 + r][l16 + 16] = f2bf(e1);
        }

        // ---- O += P * V ; rowsum += P * ones  (A = P 16x32 via wave LDS) ----
        FragB pa;
        pa.q[0] = *(const uint4*)&Pw[wave][l16][hl * 8];
        pa.q[1] = *(const uint4*)&Pw[wave][l16][hl * 8 + 16];
        #pragma unroll
        for (int f = 0; f < 4; ++f) {
            FragB vf;
            const uint4* p = (const uint4*)&Vt[buf][f * 16 + l16][hl * 8];
            vf.q[0] = p[0];
            vf.q[1] = p[2];
            acc[f] = __builtin_amdgcn_wmma_f32_16x16x32_bf16(
                         false, pa.v, false, vf.v, (short)0, acc[f], false, false);
        }
        acc_l = __builtin_amdgcn_wmma_f32_16x16x32_bf16(
                    false, pa.v, false, ones.v, (short)0, acc_l, false, false);

        // ---- convert + store next tile into the other buffer ----
        if (have_next) store_tile(buf ^ 1, krn, vrn, mrn);
        __syncthreads();   // next-buffer stores visible; current buf free to reuse
    }

    // ---- epilogue: O = acc / rowsum (rowsum lives in lane hl*16, col 0) ----
    #pragma unroll
    for (int r = 0; r < 8; ++r) {
        const int qr = q_base + hl * 8 + r;
        const float lsum = __shfl(acc_l[r], hl * 16);
        const float inv  = 1.0f / lsum;
        float* orow = Ob + (size_t)qr * D_DIM;
        #pragma unroll
        for (int f = 0; f < 4; ++f)
            orow[f * 16 + l16] = acc[f][r] * inv;
    }
}

extern "C" void kernel_launch(void* const* d_in, const int* in_sizes, int n_in,
                              void* d_out, int out_size, void* d_ws, size_t ws_size,
                              hipStream_t stream) {
    (void)in_sizes; (void)n_in; (void)out_size; (void)d_ws; (void)ws_size;
    const float* Q = (const float*)d_in[0];
    const float* K = (const float*)d_in[1];
    const float* V = (const float*)d_in[2];
    const unsigned char* mask = (const unsigned char*)d_in[3];
    float* out = (float*)d_out;

    dim3 grid(S_LEN / BLOCK_M, BH);   // (32, 32) blocks
    dim3 block(128);                  // 4 waves (wave32)
    fa_fwd_kernel<<<grid, block, 0, stream>>>(Q, K, V, mask, out);
}
